// MultiHeadAttention_21723944583578
// MI455X (gfx1250) — compile-verified, matrix operand order checked by reading, not run on hardware
//
#include <hip/hip_runtime.h>
#include <hip/hip_bf16.h>
#include <stdint.h>

// ---------------------------------------------------------------------------
// MI455X (gfx1250) MHA forward: bf16 WMMA, 2x2 tiles per wave, async-to-LDS
// DMA with double buffering in the attention loop. Workspace (64 MB):
//   [0,16M)  Qp  bf16 [B,H,S,DH]
//   [16,32M) Kp  bf16 [B,H,S,DH]
//   [32,48M) VpT bf16 [B,H,DH,S]
//   [48,64M) Cc  bf16 [B,S,H*DH]
// ---------------------------------------------------------------------------

typedef __bf16 bf16_t;
typedef __attribute__((ext_vector_type(16))) __bf16 v16bf;
typedef __attribute__((ext_vector_type(4)))  __bf16 v4bf;
typedef __attribute__((ext_vector_type(2)))  __bf16 v2bf;
typedef __attribute__((ext_vector_type(8)))  float  v8f;
typedef __attribute__((ext_vector_type(4)))  float  v4f;

constexpr int Bz = 4, Sz = 2048, Dz = 1024, Hz = 16, DHz = 64;
constexpr int LDT = 80;   // bf16 elems per LDS row = 160 B (16B-aligned rows)

// --- WMMA fragment loader (ISA 7.12.2 16-bit layouts) ----------------------
// A (16x32 MxK): lane L holds row M=L%16; frag elems 0..7 are K=(L/16)*8..+7,
// elems 8..15 are K=16+(L/16)*8..+7 -> two ds_load_b128 per fragment.
// B (32x16 KxN): same pattern, tile stored n-major (elem(k,n)=base[n*ld+k]).
__device__ inline v16bf load_frag(const bf16_t* base, int ld, int rn0, int k0, int lane) {
  const int rn   = rn0 + (lane & 15);
  const int half = (lane >> 4) & 1;
  const bf16_t* p = base + rn * ld + k0 + half * 8;
  v16bf r;
  ((uint4*)&r)[0] = *(const uint4*)(p);
  ((uint4*)&r)[1] = *(const uint4*)(p + 16);
  return r;
}

__device__ inline v8f wmma_bf16(v16bf a, v16bf b, v8f c) {
  return __builtin_amdgcn_wmma_f32_16x16x32_bf16(false, a, false, b, (short)0, c,
                                                 false, false);
}

// --- CDNA5 async global->LDS copy (16 bytes per lane) ----------------------
__device__ inline void async_copy_b128(uint32_t lds_byte_off, const void* gptr) {
  asm volatile("global_load_async_to_lds_b128 %0, %1, off"
               :: "v"(lds_byte_off), "v"(gptr) : "memory");
}

// --- staging helpers -------------------------------------------------------
// row-major fp32 -> bf16 LDS tile: one float4 -> one ds_store_b64
__device__ inline void stage_row4(bf16_t* dst, const float* src) {
  const v4f f = *(const v4f*)(src);
  *(v4bf*)(dst) = __builtin_convertvector(f, v4bf);
}
// n-major (transposed) W tile staging with d-pair packing: thread covers a
// 2(d) x 4(e) block -> four ds_store_b32.
__device__ inline void stage_w_pair(bf16_t* Wl, const float* W, int ldw,
                                    int d0, int e0) {
  const v4f f0 = *(const v4f*)(W + (size_t)d0 * ldw + e0);
  const v4f f1 = *(const v4f*)(W + (size_t)(d0 + 1) * ldw + e0);
#pragma unroll
  for (int j = 0; j < 4; ++j) {
    v2bf pk = { (bf16_t)f0[j], (bf16_t)f1[j] };   // v_cvt_pk_bf16_f32
    *(v2bf*)(&Wl[(e0 + j) * LDT + d0]) = pk;
  }
}

// ===========================================================================
// Kernel 1: per-head projection. Block tile 128(s) x 64(e), 8 waves in 4x2,
// each wave owns 2x2 16x16 tiles (4 accs, 4 WMMA per 32-K step).
// transpose_out==1 stores out[b,h,e,s] (for V).
// ===========================================================================
__global__ __launch_bounds__(256) void qkv_proj_kernel(
    const float* __restrict__ x, const float* __restrict__ W,
    const float* __restrict__ bias, bf16_t* __restrict__ out, int transpose_out)
{
  __shared__ bf16_t Al[128 * LDT];  // m-major x tile (128 x 64)
  __shared__ bf16_t Wl[64 * LDT];   // n-major W tile (64e x 64d)
  const int tid = threadIdx.x, lane = tid & 31, wv = tid >> 5;
  const int mi = wv >> 1, ni = wv & 1;           // 4 x 2 wave grid
  const int s0 = blockIdx.x * 128, h = blockIdx.y, b = blockIdx.z;
  const float* xb = x + ((size_t)b * Sz + s0) * Dz;
  const float* Wh = W + (size_t)h * Dz * DHz;
  v8f acc[2][2] = {};
  for (int dk = 0; dk < Dz; dk += 64) {
    // stage A: 128x64 fp32 -> bf16 (2048 float4, 8 per thread, b64 stores)
#pragma unroll
    for (int it = 0; it < 8; ++it) {
      const int f4 = tid + it * 256;
      const int r = f4 >> 4, c = (f4 & 15) * 4;
      stage_row4(&Al[r * LDT + c], xb + (size_t)r * Dz + dk + c);
    }
    // stage W: 64x64 fp32 -> bf16 transposed, packed pair stores
#pragma unroll
    for (int it = 0; it < 2; ++it) {
      const int d0 = (tid >> 4) * 2 + it * 32;
      const int e0 = (tid & 15) * 4;
      stage_w_pair(Wl, Wh + (size_t)dk * DHz, DHz, d0, e0);
    }
    __syncthreads();
#pragma unroll
    for (int k0 = 0; k0 < 64; k0 += 32) {
      v16bf a0 = load_frag(Al, LDT, mi * 32,      k0, lane);
      v16bf a1 = load_frag(Al, LDT, mi * 32 + 16, k0, lane);
      v16bf b0 = load_frag(Wl, LDT, ni * 32,      k0, lane);
      v16bf b1 = load_frag(Wl, LDT, ni * 32 + 16, k0, lane);
      acc[0][0] = wmma_bf16(a0, b0, acc[0][0]);
      acc[0][1] = wmma_bf16(a0, b1, acc[0][1]);
      acc[1][0] = wmma_bf16(a1, b0, acc[1][0]);
      acc[1][1] = wmma_bf16(a1, b1, acc[1][1]);
    }
    __syncthreads();
  }
  const int n = lane & 15, half = lane >> 4;
#pragma unroll
  for (int sm = 0; sm < 2; ++sm)
#pragma unroll
    for (int sn = 0; sn < 2; ++sn) {
      const int e = ni * 32 + sn * 16 + n;
      const float bias_e = bias[h * DHz + e];
#pragma unroll
      for (int r = 0; r < 8; ++r) {
        const int m = mi * 32 + sm * 16 + r + 8 * half;
        const float val = acc[sm][sn][r] + bias_e;
        if (!transpose_out)
          out[(((size_t)b * Hz + h) * Sz + s0 + m) * DHz + e] = (bf16_t)val;
        else
          out[(((size_t)b * Hz + h) * DHz + e) * (size_t)Sz + s0 + m] = (bf16_t)val;
      }
    }
}

// ===========================================================================
// Kernel 2: flash attention. 64-query block, 64 keys/iter, double-buffered
// K/V async-to-LDS, online softmax fp32. Wave grid 2(m)x4(n), 2 m-subtiles.
// ===========================================================================
constexpr uint32_t AOFF_Q  = 0;                        // 64x64 bf16
constexpr uint32_t AOFF_K0 = AOFF_Q  + 64 * LDT * 2;   // 10240
constexpr uint32_t AOFF_K1 = AOFF_K0 + 64 * LDT * 2;   // 20480
constexpr uint32_t AOFF_V0 = AOFF_K1 + 64 * LDT * 2;   // 30720
constexpr uint32_t AOFF_V1 = AOFF_V0 + 64 * LDT * 2;   // 40960
constexpr uint32_t AOFF_S  = AOFF_V1 + 64 * LDT * 2;   // 51200 (fp32 64x65)
constexpr uint32_t AOFF_P  = AOFF_S  + 64 * 65 * 4;    // 67840 (bf16 64xLDT)
constexpr uint32_t AOFF_AL = AOFF_P  + 64 * LDT * 2;   // 78080
constexpr uint32_t AOFF_L  = AOFF_AL + 64 * 4;         // 78336
constexpr uint32_t ATTN_SMEM = AOFF_L + 64 * 4;        // 78592 bytes

__device__ inline void attn_prefetch(const bf16_t* Kg, const bf16_t* Vg,
                                     uint32_t koff, uint32_t voff, int kb, int tid)
{
  const int row = tid >> 3, co = (tid & 7) * 8;   // 16B chunk per lane
  async_copy_b128(koff + (uint32_t)(row * LDT + co) * 2,
                  Kg + (size_t)(kb + row) * DHz + co);
  async_copy_b128(koff + (uint32_t)((row + 32) * LDT + co) * 2,
                  Kg + (size_t)(kb + row + 32) * DHz + co);
  async_copy_b128(voff + (uint32_t)(row * LDT + co) * 2,
                  Vg + (size_t)row * Sz + kb + co);
  async_copy_b128(voff + (uint32_t)((row + 32) * LDT + co) * 2,
                  Vg + (size_t)(row + 32) * Sz + kb + co);
}

__global__ __launch_bounds__(256) void attn_kernel(
    const bf16_t* __restrict__ Qp, const bf16_t* __restrict__ Kp,
    const bf16_t* __restrict__ VpT, bf16_t* __restrict__ concat)
{
  extern __shared__ char smem[];
  bf16_t* Ql = (bf16_t*)(smem + AOFF_Q);
  float*  Sb = (float*)(smem + AOFF_S);
  bf16_t* Pb = (bf16_t*)(smem + AOFF_P);
  float*  alpha_s = (float*)(smem + AOFF_AL);
  float*  lrow_s  = (float*)(smem + AOFF_L);

  const int tid = threadIdx.x, lane = tid & 31, wv = tid >> 5;
  const int mi = wv >> 2, ni = wv & 3;            // 2 x 4 wave grid
  const int s0 = blockIdx.x * 64;
  const int h = blockIdx.y, b = blockIdx.z;
  const size_t bh = (size_t)b * Hz + h;
  const bf16_t* Qg = Qp + (bh * Sz + s0) * DHz;
  const bf16_t* Kg = Kp + bh * (size_t)Sz * DHz;
  const bf16_t* Vg = VpT + bh * (size_t)DHz * Sz;

  // async-load Q tile (64x64 bf16): two b128 per thread
#pragma unroll
  for (int it = 0; it < 2; ++it) {
    const int ci = tid + it * 256;
    const int row = ci >> 3, co = (ci & 7) * 8;
    async_copy_b128(AOFF_Q + (uint32_t)(row * LDT + co) * 2,
                    Qg + (size_t)row * DHz + co);
  }
  attn_prefetch(Kg, Vg, AOFF_K0, AOFF_V0, 0, tid);

  float m_r = -__builtin_inff(), l_r = 0.f;       // live in threads 0..63
  v8f accO0 = {}, accO1 = {};
  const int n = lane & 15, half = lane >> 4;
  constexpr int NKB = Sz / 64;

  for (int ib = 0; ib < NKB; ++ib) {
    const int cur = ib & 1;
    const bf16_t* Kc = (bf16_t*)(smem + (cur ? AOFF_K1 : AOFF_K0));
    const bf16_t* Vc = (bf16_t*)(smem + (cur ? AOFF_V1 : AOFF_V0));
    if (ib + 1 < NKB) {
      attn_prefetch(Kg, Vg, cur ? AOFF_K0 : AOFF_K1, cur ? AOFF_V0 : AOFF_V1,
                    (ib + 1) * 64, tid);
      asm volatile("s_wait_asynccnt 4" ::: "memory");
    } else {
      asm volatile("s_wait_asynccnt 0" ::: "memory");
    }
    __syncthreads();

    // scores S = (Q K^T) * 1/sqrt(DH) : 2 m-subtiles per wave
    v8f sc0 = {}, sc1 = {};
#pragma unroll
    for (int k0 = 0; k0 < 64; k0 += 32) {
      v16bf a0 = load_frag(Ql, LDT, mi * 32,      k0, lane);
      v16bf a1 = load_frag(Ql, LDT, mi * 32 + 16, k0, lane);
      v16bf bb = load_frag(Kc, LDT, ni * 16,      k0, lane);
      sc0 = wmma_bf16(a0, bb, sc0);
      sc1 = wmma_bf16(a1, bb, sc1);
    }
#pragma unroll
    for (int r = 0; r < 8; ++r) {
      Sb[(mi * 32 + r + 8 * half) * 65 + ni * 16 + n]      = sc0[r] * 0.125f;
      Sb[(mi * 32 + 16 + r + 8 * half) * 65 + ni * 16 + n] = sc1[r] * 0.125f;
    }
    __syncthreads();

    // online softmax: waves 0-1, one row per thread, packed bf16 P stores
    if (tid < 64) {
      float mx = m_r;
#pragma unroll
      for (int j = 0; j < 64; ++j) mx = fmaxf(mx, Sb[tid * 65 + j]);
      const float aa = __expf(m_r - mx);
      float sum = 0.f;
#pragma unroll
      for (int j0 = 0; j0 < 64; j0 += 4) {
        v4f p;
#pragma unroll
        for (int j = 0; j < 4; ++j) {
          p[j] = __expf(Sb[tid * 65 + j0 + j] - mx);
          sum += p[j];
        }
        *(v4bf*)(&Pb[tid * LDT + j0]) = __builtin_convertvector(p, v4bf);
      }
      l_r = l_r * aa + sum;
      m_r = mx;
      alpha_s[tid] = aa;
      if (ib == NKB - 1) lrow_s[tid] = l_r;
    }
    __syncthreads();

    // rescale running O, then O += P * V
#pragma unroll
    for (int r = 0; r < 8; ++r) {
      accO0[r] *= alpha_s[mi * 32 + r + 8 * half];
      accO1[r] *= alpha_s[mi * 32 + 16 + r + 8 * half];
    }
#pragma unroll
    for (int k0 = 0; k0 < 64; k0 += 32) {
      v16bf a0 = load_frag(Pb, LDT, mi * 32,      k0, lane);
      v16bf a1 = load_frag(Pb, LDT, mi * 32 + 16, k0, lane);
      v16bf bb = load_frag(Vc, LDT, ni * 16,      k0, lane);
      accO0 = wmma_bf16(a0, bb, accO0);
      accO1 = wmma_bf16(a1, bb, accO1);
    }
    __syncthreads();   // all reads of K/V/P done before next async overwrite
  }

#pragma unroll
  for (int r = 0; r < 8; ++r) {
    const int m0 = mi * 32 + r + 8 * half;
    const int m1 = m0 + 16;
    const float o0 = accO0[r] / lrow_s[m0];
    const float o1 = accO1[r] / lrow_s[m1];
    const size_t col = (size_t)h * DHz + ni * 16 + n;
    concat[((size_t)b * Sz + s0 + m0) * (Hz * DHz) + col] = (bf16_t)o0;
    concat[((size_t)b * Sz + s0 + m1) * (Hz * DHz) + col] = (bf16_t)o1;
  }
}

// ===========================================================================
// Kernel 3: out = concat[8192,1024](bf16) @ Wo[1024,1024] + bo (fp32 out).
// Block tile 128(M) x 64(N), wave grid 4x2, 2x2 tiles per wave; A via async.
// ===========================================================================
constexpr uint32_t OOFF_A = 0;                           // 128x64 bf16
constexpr uint32_t OOFF_W = OOFF_A + 128 * LDT * 2;      // 20480
constexpr uint32_t OPROJ_SMEM = OOFF_W + 64 * LDT * 2;   // 30720 bytes

__global__ __launch_bounds__(256) void out_proj_kernel(
    const bf16_t* __restrict__ X, const float* __restrict__ Wo,
    const float* __restrict__ bo, float* __restrict__ out)
{
  extern __shared__ char smem[];
  bf16_t* Al = (bf16_t*)(smem + OOFF_A);
  bf16_t* Wl = (bf16_t*)(smem + OOFF_W);   // n-major
  const int tid = threadIdx.x, lane = tid & 31, wv = tid >> 5;
  const int mi = wv >> 1, ni = wv & 1;     // 4 x 2 wave grid
  const int r0 = blockIdx.x * 128, c0 = blockIdx.y * 64;
  v8f acc[2][2] = {};
  for (int dk = 0; dk < Hz * DHz; dk += 64) {
    // A tile (bf16): async DMA, four b128 per thread
#pragma unroll
    for (int it = 0; it < 4; ++it) {
      const int ci = tid + it * 256;
      const int row = ci >> 3, co = (ci & 7) * 8;
      async_copy_b128(OOFF_A + (uint32_t)(row * LDT + co) * 2,
                      X + (size_t)(r0 + row) * (Hz * DHz) + dk + co);
    }
    // W tile: fp32 -> bf16 transposed, packed pair stores
#pragma unroll
    for (int it = 0; it < 2; ++it) {
      const int d0 = (tid >> 4) * 2 + it * 32;
      const int e0 = (tid & 15) * 4;
      stage_w_pair(Wl, Wo + (size_t)dk * Dz + c0, Dz, d0, e0);
    }
    asm volatile("s_wait_asynccnt 0" ::: "memory");
    __syncthreads();
#pragma unroll
    for (int k0 = 0; k0 < 64; k0 += 32) {
      v16bf a0 = load_frag(Al, LDT, mi * 32,      k0, lane);
      v16bf a1 = load_frag(Al, LDT, mi * 32 + 16, k0, lane);
      v16bf b0 = load_frag(Wl, LDT, ni * 32,      k0, lane);
      v16bf b1 = load_frag(Wl, LDT, ni * 32 + 16, k0, lane);
      acc[0][0] = wmma_bf16(a0, b0, acc[0][0]);
      acc[0][1] = wmma_bf16(a0, b1, acc[0][1]);
      acc[1][0] = wmma_bf16(a1, b0, acc[1][0]);
      acc[1][1] = wmma_bf16(a1, b1, acc[1][1]);
    }
    __syncthreads();
  }
  const int n = lane & 15, half = lane >> 4;
#pragma unroll
  for (int sm = 0; sm < 2; ++sm)
#pragma unroll
    for (int sn = 0; sn < 2; ++sn) {
      const int e = c0 + ni * 32 + sn * 16 + n;
      const float bias_e = bo[e];
#pragma unroll
      for (int r = 0; r < 8; ++r)
        out[(size_t)(r0 + mi * 32 + sm * 16 + r + 8 * half) * Dz + e] =
            acc[sm][sn][r] + bias_e;
    }
}

// ===========================================================================
extern "C" void kernel_launch(void* const* d_in, const int* in_sizes, int n_in,
                              void* d_out, int out_size, void* d_ws, size_t ws_size,
                              hipStream_t stream)
{
  const float* q  = (const float*)d_in[0];
  const float* k  = (const float*)d_in[1];
  const float* v  = (const float*)d_in[2];
  const float* Wq = (const float*)d_in[3];
  const float* bq = (const float*)d_in[4];
  const float* Wk = (const float*)d_in[5];
  const float* bk = (const float*)d_in[6];
  const float* Wv = (const float*)d_in[7];
  const float* bvv = (const float*)d_in[8];
  const float* Wo = (const float*)d_in[9];
  const float* bo = (const float*)d_in[10];
  float* out = (float*)d_out;

  const size_t headSz = (size_t)Bz * Hz * Sz * DHz * sizeof(bf16_t);  // 16 MB
  char* ws = (char*)d_ws;
  bf16_t* Qp  = (bf16_t*)(ws);
  bf16_t* Kp  = (bf16_t*)(ws + headSz);
  bf16_t* VpT = (bf16_t*)(ws + 2 * headSz);
  bf16_t* Cc  = (bf16_t*)(ws + 3 * headSz);

  dim3 gProj(Sz / 128, Hz, Bz);
  qkv_proj_kernel<<<gProj, 256, 0, stream>>>(q, Wq, bq, Qp, 0);
  qkv_proj_kernel<<<gProj, 256, 0, stream>>>(k, Wk, bk, Kp, 0);
  qkv_proj_kernel<<<gProj, 256, 0, stream>>>(v, Wv, bvv, VpT, 1);

  dim3 gAttn(Sz / 64, Hz, Bz);
  attn_kernel<<<gAttn, 256, ATTN_SMEM, stream>>>(Qp, Kp, VpT, Cc);

  dim3 gOut((Bz * Sz) / 128, Dz / 64);
  out_proj_kernel<<<gOut, 256, OPROJ_SMEM, stream>>>(Cc, Wo, bo, out);
}
